// EncoderBlock_88038239633589
// MI455X (gfx1250) — compile-verified, matrix operand order checked by reading, not run on hardware
//
#include <hip/hip_runtime.h>

// ---------------------------------------------------------------------------
// Types for CDNA5 WMMA f16 path
// ---------------------------------------------------------------------------
typedef _Float16 h16;
typedef __attribute__((ext_vector_type(16))) _Float16 v16h;
typedef __attribute__((ext_vector_type(8)))  _Float16 v8h;
typedef __attribute__((ext_vector_type(8)))  float    v8f;

union V16 { v16h v; v8h h[2]; };

static __device__ __forceinline__ v8f wmma_f16(v16h a, v16h b, v8f c) {
    return __builtin_amdgcn_wmma_f32_16x16x32_f16(false, a, false, b,
                                                  (short)0, c, false, false);
}

// ---------------------------------------------------------------------------
// fp32 -> f16 conversion (weights)
// ---------------------------------------------------------------------------
__global__ __launch_bounds__(256)
void enc_cvt_f32_f16(const float* __restrict__ in, h16* __restrict__ out, int n) {
    int i = blockIdx.x * 256 + threadIdx.x;
    if (i < n) out[i] = (h16)in[i];
}

// ---------------------------------------------------------------------------
// LayerNorm (unbiased variance, ddof=1), fp32 in -> f16 out
// One block (256 threads) per row of D elements.
// ---------------------------------------------------------------------------
__global__ __launch_bounds__(256)
void enc_layernorm_f16(const float* __restrict__ x, const float* __restrict__ alpha,
                       const float* __restrict__ beta, h16* __restrict__ out, int D) {
    int row = blockIdx.x;
    const float* xr = x + (size_t)row * D;
    float s = 0.f, sq = 0.f;
    for (int c = threadIdx.x; c < D; c += 256) {
        float v = xr[c]; s += v; sq += v * v;
    }
    #pragma unroll
    for (int o = 16; o > 0; o >>= 1) {
        s  += __shfl_down(s,  o, 32);
        sq += __shfl_down(sq, o, 32);
    }
    __shared__ float rs[8], rq[8];
    __shared__ float mean_s, rstd_s;
    int lane = threadIdx.x & 31, wave = threadIdx.x >> 5;
    if (lane == 0) { rs[wave] = s; rq[wave] = sq; }
    __syncthreads();
    if (threadIdx.x == 0) {
        float S0 = 0.f, Q0 = 0.f;
        #pragma unroll
        for (int i = 0; i < 8; ++i) { S0 += rs[i]; Q0 += rq[i]; }
        float mean = S0 / (float)D;
        float var  = (Q0 - S0 * S0 / (float)D) / (float)(D - 1);
        mean_s = mean;
        rstd_s = rsqrtf(var + 1e-6f);
    }
    __syncthreads();
    float mean = mean_s, rstd = rstd_s;
    for (int c = threadIdx.x; c < D; c += 256) {
        float v = alpha[c] * (xr[c] - mean) * rstd + beta[c];
        out[(size_t)row * D + c] = (h16)v;
    }
}

// ---------------------------------------------------------------------------
// Tiled WMMA GEMM:  C[M,N] = A[M,K](f16) @ B[K,N](f16) + bias
// Template-specialized epilogue (branch-free): ReLU / fp32 residual /
// f32 and/or f16 output. Double-buffered LDS with register prefetch so
// global_load_b128 of tile k+1 overlaps the 8 WMMAs of tile k; one
// barrier per K-step (ping-pong buffers make the second barrier redundant).
// Block = 256 threads (8 waves), tile 128x128, BK = 32 (one WMMA K-step).
// Requires M%128==0, N%128==0, K%32==0 (true for all call sites here).
// ---------------------------------------------------------------------------
#define GBM 128
#define GBN 128
#define GBK 32
#define GPAD 40   // LDS row stride in halves (padded, 16B-aligned)

template<bool RELU, bool RESID, bool OUTF, bool OUTH>
__global__ __launch_bounds__(256)
void enc_gemm_wmma(const h16* __restrict__ A, const h16* __restrict__ Bm,
                   const float* __restrict__ bias, const float* __restrict__ resid,
                   float* __restrict__ outF, h16* __restrict__ outH,
                   int M, int N, int K) {
    __shared__ h16 sA [2][GBM * GPAD];   // [buf][row][k]
    __shared__ h16 sBt[2][GBN * GPAD];   // [buf][col][k]  (B transposed)

    int tid  = threadIdx.x;
    int lane = tid & 31, wave = tid >> 5;
    int hlf  = lane >> 4, l16 = lane & 15;
    int wm   = (wave & 1) * 64;       // wave M offset in tile
    int wn   = (wave >> 1) * 32;      // wave N offset in tile
    int bm0  = blockIdx.y * GBM;
    int bn0  = blockIdx.x * GBN;

    // Per-thread staging assignments (fixed across iterations).
    int ca  = tid * 2;                // A chunks: row = c>>2, col = (c&3)*8
    int arow0 = (ca)     >> 2, acol0 = ((ca)     & 3) * 8;
    int arow1 = (ca + 1) >> 2, acol1 = ((ca + 1) & 3) * 8;
    int bkr0  = (ca)     >> 4, bcol0 = ((ca)     & 15) * 8;
    int bkr1  = (ca + 1) >> 4, bcol1 = ((ca + 1) & 15) * 8;

    v8f acc[4][2];
    #pragma unroll
    for (int ms = 0; ms < 4; ++ms)
        #pragma unroll
        for (int ns = 0; ns < 2; ++ns)
            #pragma unroll
            for (int r = 0; r < 8; ++r) acc[ms][ns][r] = 0.f;

    const int NT = K / GBK;
    v8h ra0, ra1, rb0, rb1;

    // Prefetch tile 0 into registers.
    ra0 = *(const v8h*)(A  + (size_t)(bm0 + arow0) * K + acol0);
    ra1 = *(const v8h*)(A  + (size_t)(bm0 + arow1) * K + acol1);
    rb0 = *(const v8h*)(Bm + (size_t)(bkr0)        * N + bn0 + bcol0);
    rb1 = *(const v8h*)(Bm + (size_t)(bkr1)        * N + bn0 + bcol1);

    int buf = 0;
    for (int kt = 0; kt < NT; ++kt) {
        // Commit prefetched registers to LDS buffer `buf`.
        *(v8h*)(sA[buf] + arow0 * GPAD + acol0) = ra0;
        *(v8h*)(sA[buf] + arow1 * GPAD + acol1) = ra1;
        #pragma unroll
        for (int j = 0; j < 8; ++j) sBt[buf][(bcol0 + j) * GPAD + bkr0] = rb0[j];
        #pragma unroll
        for (int j = 0; j < 8; ++j) sBt[buf][(bcol1 + j) * GPAD + bkr1] = rb1[j];
        __syncthreads();

        // Issue global prefetch of tile kt+1 (overlaps WMMA below).
        if (kt + 1 < NT) {
            int k0 = (kt + 1) * GBK;
            ra0 = *(const v8h*)(A  + (size_t)(bm0 + arow0) * K + k0 + acol0);
            ra1 = *(const v8h*)(A  + (size_t)(bm0 + arow1) * K + k0 + acol1);
            rb0 = *(const v8h*)(Bm + (size_t)(k0 + bkr0)   * N + bn0 + bcol0);
            rb1 = *(const v8h*)(Bm + (size_t)(k0 + bkr1)   * N + bn0 + bcol1);
        }

        // Fragments: A lane=row, halves K={hlf*8+0..7, 16+hlf*8+0..7}; B mirrored.
        V16 af[4], bf[2];
        #pragma unroll
        for (int ms = 0; ms < 4; ++ms) {
            const h16* p = sA[buf] + (wm + ms * 16 + l16) * GPAD;
            af[ms].h[0] = *(const v8h*)(p + hlf * 8);
            af[ms].h[1] = *(const v8h*)(p + 16 + hlf * 8);
        }
        #pragma unroll
        for (int ns = 0; ns < 2; ++ns) {
            const h16* p = sBt[buf] + (wn + ns * 16 + l16) * GPAD;
            bf[ns].h[0] = *(const v8h*)(p + hlf * 8);
            bf[ns].h[1] = *(const v8h*)(p + 16 + hlf * 8);
        }
        #pragma unroll
        for (int ms = 0; ms < 4; ++ms)
            #pragma unroll
            for (int ns = 0; ns < 2; ++ns)
                acc[ms][ns] = wmma_f16(af[ms].v, bf[ns].v, acc[ms][ns]);

        buf ^= 1;
        // No second barrier: iteration kt+2 reuses this buffer only after the
        // kt+1 barrier, which every wave reaches after finishing compute(kt).
    }

    // Branch-free epilogue: row = tile + r + 8*hlf, col = tile + l16.
    #pragma unroll
    for (int ms = 0; ms < 4; ++ms)
        #pragma unroll
        for (int ns = 0; ns < 2; ++ns) {
            int col  = bn0 + wn + ns * 16 + l16;
            float bv = bias[col];
            int rowb = bm0 + wm + ms * 16 + hlf * 8;
            float res[8];
            if (RESID) {
                #pragma unroll
                for (int r = 0; r < 8; ++r)
                    res[r] = resid[(size_t)(rowb + r) * N + col];
            }
            #pragma unroll
            for (int r = 0; r < 8; ++r) {
                float v = acc[ms][ns][r] + bv;
                if (RELU) v = v > 0.f ? v : 0.f;
                if (RESID) v += res[r];
                size_t idx = (size_t)(rowb + r) * N + col;
                if (OUTF) outF[idx] = v;
                if (OUTH) outH[idx] = (h16)v;
            }
        }
}

// ---------------------------------------------------------------------------
// Flash attention: one block = 64 query rows of one (b,h); 4 waves, 16 rows
// each. Streams keys in 64-wide tiles held in LDS; online softmax; both
// QK^T and P@V via v_wmma_f32_16x16x32_f16. Q/K/V/ctx layout: [B*S, H*64].
// ---------------------------------------------------------------------------
#define SPAD 72

__global__ __launch_bounds__(128)
void enc_flash_attn(const h16* __restrict__ q, const h16* __restrict__ kk,
                    const h16* __restrict__ vv, const int* __restrict__ mask,
                    h16* __restrict__ ctx, int S) {
    __shared__ h16 sK [64 * SPAD];      // [key][dim]
    __shared__ h16 sVt[64 * SPAD];      // [dim][key]  (V transposed)
    __shared__ h16 sP [4][16 * SPAD];   // per-wave P transpose staging

    const int DM = 1024;
    int tid  = threadIdx.x;
    int lane = tid & 31, wave = tid >> 5;
    int hlf  = lane >> 4, l16 = lane & 15;
    int qt   = blockIdx.x;              // q tile (64 rows)
    int h    = blockIdx.y;
    int b    = blockIdx.z;
    int ho   = h * 64;

    // Q fragments for this wave's 16 rows (A layout, direct from global).
    int qtok = b * S + qt * 64 + wave * 16 + l16;
    V16 aq[2];
    #pragma unroll
    for (int kc = 0; kc < 2; ++kc) {
        const h16* p = q + (size_t)qtok * DM + ho + kc * 32;
        aq[kc].h[0] = *(const v8h*)(p + hlf * 8);
        aq[kc].h[1] = *(const v8h*)(p + 16 + hlf * 8);
    }

    v8f   o[4];
    float m8[8], l8[8];
    #pragma unroll
    for (int ns = 0; ns < 4; ++ns)
        #pragma unroll
        for (int r = 0; r < 8; ++r) o[ns][r] = 0.f;
    #pragma unroll
    for (int r = 0; r < 8; ++r) { m8[r] = -1e30f; l8[r] = 0.f; }

    for (int kt = 0; kt < S / 64; ++kt) {
        __syncthreads();
        // Stage K [key][dim] and V^T [dim][key]: 64x64 halves each.
        #pragma unroll
        for (int i = 0; i < 4; ++i) {
            int c = tid * 4 + i;
            int row = c >> 3, col = (c & 7) * 8;
            int tok = b * S + kt * 64 + row;
            v8h dk = *(const v8h*)(kk + (size_t)tok * DM + ho + col);
            *(v8h*)(sK + row * SPAD + col) = dk;
            v8h dv = *(const v8h*)(vv + (size_t)tok * DM + ho + col);
            #pragma unroll
            for (int j = 0; j < 8; ++j) sVt[(col + j) * SPAD + row] = dv[j];
        }
        __syncthreads();

        // Scores: 16(q) x 64(keys), Dk=64 => 2 WMMA K-steps per 16-key block.
        v8f sacc[4];
        #pragma unroll
        for (int ns = 0; ns < 4; ++ns) {
            #pragma unroll
            for (int r = 0; r < 8; ++r) sacc[ns][r] = 0.f;
            #pragma unroll
            for (int kc = 0; kc < 2; ++kc) {
                V16 bk;
                const h16* p = sK + (ns * 16 + l16) * SPAD + kc * 32;
                bk.h[0] = *(const v8h*)(p + hlf * 8);
                bk.h[1] = *(const v8h*)(p + 16 + hlf * 8);
                sacc[ns] = wmma_f16(aq[kc].v, bk.v, sacc[ns]);
            }
        }
        // Scale 1/sqrt(64) and mask (this lane's column is one key).
        #pragma unroll
        for (int ns = 0; ns < 4; ++ns) {
            int key = kt * 64 + ns * 16 + l16;
            int mk  = mask[b * S + key];
            #pragma unroll
            for (int r = 0; r < 8; ++r)
                sacc[ns][r] = mk ? sacc[ns][r] * 0.125f : -1e9f;
        }
        // Row max across 64 columns (cross-lane within each 16-lane group).
        float curm[8];
        #pragma unroll
        for (int r = 0; r < 8; ++r) {
            float mv = sacc[0][r];
            #pragma unroll
            for (int ns = 1; ns < 4; ++ns) mv = fmaxf(mv, sacc[ns][r]);
            #pragma unroll
            for (int d = 1; d < 16; d <<= 1) mv = fmaxf(mv, __shfl_xor(mv, d, 32));
            curm[r] = mv;
        }
        float scl[8], curl[8];
        #pragma unroll
        for (int r = 0; r < 8; ++r) {
            float nm = fmaxf(m8[r], curm[r]);
            scl[r] = __expf(m8[r] - nm);
            m8[r]  = nm;
            curl[r] = 0.f;
        }
        #pragma unroll
        for (int ns = 0; ns < 4; ++ns)
            #pragma unroll
            for (int r = 0; r < 8; ++r) {
                float pv = __expf(sacc[ns][r] - m8[r]);
                sacc[ns][r] = pv;
                curl[r] += pv;
            }
        #pragma unroll
        for (int r = 0; r < 8; ++r) {
            float sv = curl[r];
            #pragma unroll
            for (int d = 1; d < 16; d <<= 1) sv += __shfl_xor(sv, d, 32);
            l8[r] = l8[r] * scl[r] + sv;
            #pragma unroll
            for (int ns = 0; ns < 4; ++ns) o[ns][r] *= scl[r];
        }
        // Transpose P through LDS (C layout -> A layout): row = r + 8*hlf.
        #pragma unroll
        for (int ns = 0; ns < 4; ++ns) {
            int col = ns * 16 + l16;
            #pragma unroll
            for (int r = 0; r < 8; ++r)
                sP[wave][(r + 8 * hlf) * SPAD + col] = (h16)sacc[ns][r];
        }
        __syncthreads();
        // O += P @ V  (A = P 16x32 key-chunks, B = V^T fragments).
        #pragma unroll
        for (int kc = 0; kc < 2; ++kc) {
            V16 ap;
            const h16* pp = &sP[wave][l16 * SPAD + kc * 32];
            ap.h[0] = *(const v8h*)(pp + hlf * 8);
            ap.h[1] = *(const v8h*)(pp + 16 + hlf * 8);
            #pragma unroll
            for (int ns = 0; ns < 4; ++ns) {
                V16 bv;
                const h16* pv = sVt + (ns * 16 + l16) * SPAD + kc * 32;
                bv.h[0] = *(const v8h*)(pv + hlf * 8);
                bv.h[1] = *(const v8h*)(pv + 16 + hlf * 8);
                o[ns] = wmma_f16(ap.v, bv.v, o[ns]);
            }
        }
    }
    // Normalize and store ctx (f16, same [B*S, H*64] layout).
    #pragma unroll
    for (int ns = 0; ns < 4; ++ns) {
        int col = ho + ns * 16 + l16;
        #pragma unroll
        for (int r = 0; r < 8; ++r) {
            int tok = b * S + qt * 64 + wave * 16 + r + 8 * hlf;
            ctx[(size_t)tok * DM + col] = (h16)(o[ns][r] / l8[r]);
        }
    }
}

// ---------------------------------------------------------------------------
// Host-side orchestration
// ---------------------------------------------------------------------------
extern "C" void kernel_launch(void* const* d_in, const int* in_sizes, int n_in,
                              void* d_out, int out_size, void* d_ws, size_t ws_size,
                              hipStream_t stream) {
    (void)in_sizes; (void)n_in; (void)out_size; (void)ws_size;

    const int B_ = 2, S_ = 2048, DM = 1024, DF = 4096;
    const int M = B_ * S_;  // 4096 token rows

    const float* x     = (const float*)d_in[0];
    const int*   mask  = (const int*)  d_in[1];
    const float* wq    = (const float*)d_in[2];
    const float* bq    = (const float*)d_in[3];
    const float* wk    = (const float*)d_in[4];
    const float* bk    = (const float*)d_in[5];
    const float* wv    = (const float*)d_in[6];
    const float* bv    = (const float*)d_in[7];
    const float* wo    = (const float*)d_in[8];
    const float* bo    = (const float*)d_in[9];
    const float* ln1a  = (const float*)d_in[10];
    const float* ln1b  = (const float*)d_in[11];
    const float* ln2a  = (const float*)d_in[12];
    const float* ln2b  = (const float*)d_in[13];
    const float* w1    = (const float*)d_in[14];
    const float* b1    = (const float*)d_in[15];
    const float* w2    = (const float*)d_in[16];
    const float* b2    = (const float*)d_in[17];
    float* out = (float*)d_out;

    // Workspace carve-up (~120 MB total).
    char* wp = (char*)d_ws;
    auto carve = [&](size_t bytes) { char* p = wp; wp += bytes; return p; };
    const size_t szW  = (size_t)DM * DM * sizeof(h16);   // 2 MB
    const size_t szWF = (size_t)DM * DF * sizeof(h16);   // 8 MB
    const size_t szA  = (size_t)M  * DM * sizeof(h16);   // 8 MB
    const size_t szAF = (size_t)M  * DM * sizeof(float); // 16 MB
    const size_t szH  = (size_t)M  * DF * sizeof(h16);   // 32 MB

    h16*   wq16 = (h16*)carve(szW);
    h16*   wk16 = (h16*)carve(szW);
    h16*   wv16 = (h16*)carve(szW);
    h16*   wo16 = (h16*)carve(szW);
    h16*   w116 = (h16*)carve(szWF);
    h16*   w216 = (h16*)carve(szWF);
    h16*   n1   = (h16*)carve(szA);
    h16*   q16  = (h16*)carve(szA);
    h16*   k16  = (h16*)carve(szA);
    h16*   v16  = (h16*)carve(szA);
    h16*   ctx  = (h16*)carve(szA);
    float* y1   = (float*)carve(szAF);
    h16*   n2   = (h16*)carve(szA);
    h16*   hbuf = (h16*)carve(szH);

    // 1) Weight conversion fp32 -> f16.
    auto cvt = [&](const float* src, h16* dst, int n) {
        enc_cvt_f32_f16<<<(n + 255) / 256, 256, 0, stream>>>(src, dst, n);
    };
    cvt(wq, wq16, DM * DM);
    cvt(wk, wk16, DM * DM);
    cvt(wv, wv16, DM * DM);
    cvt(wo, wo16, DM * DM);
    cvt(w1, w116, DM * DF);
    cvt(w2, w216, DM * DF);

    // 2) LN1: x -> n1 (f16).
    enc_layernorm_f16<<<M, 256, 0, stream>>>(x, ln1a, ln1b, n1, DM);

    // 3) QKV projections (WMMA GEMM, f16 out).
    dim3 gQKV(DM / GBN, M / GBM);
    enc_gemm_wmma<false, false, false, true><<<gQKV, 256, 0, stream>>>(
        n1, wq16, bq, nullptr, nullptr, q16, M, DM, DM);
    enc_gemm_wmma<false, false, false, true><<<gQKV, 256, 0, stream>>>(
        n1, wk16, bk, nullptr, nullptr, k16, M, DM, DM);
    enc_gemm_wmma<false, false, false, true><<<gQKV, 256, 0, stream>>>(
        n1, wv16, bv, nullptr, nullptr, v16, M, DM, DM);

    // 4) Flash attention: grid (q-tiles, heads, batch).
    dim3 gFA(S_ / 64, 16, B_);
    enc_flash_attn<<<gFA, 128, 0, stream>>>(q16, k16, v16, mask, ctx, S_);

    // 5) O-projection + residual: y1 = x + ctx @ wo + bo  (f32 out).
    enc_gemm_wmma<false, true, true, false><<<gQKV, 256, 0, stream>>>(
        ctx, wo16, bo, x, y1, nullptr, M, DM, DM);

    // 6) LN2: y1 -> n2 (f16).
    enc_layernorm_f16<<<M, 256, 0, stream>>>(y1, ln2a, ln2b, n2, DM);

    // 7) FFN1 + ReLU: hbuf = relu(n2 @ w1 + b1)  (f16 out).
    dim3 gF1(DF / GBN, M / GBM);
    enc_gemm_wmma<true, false, false, true><<<gF1, 256, 0, stream>>>(
        n2, w116, b1, nullptr, nullptr, hbuf, M, DF, DM);

    // 8) FFN2 + residual: out = y1 + hbuf @ w2 + b2  (f32 out to d_out).
    dim3 gF2(DM / GBN, M / GBM);
    enc_gemm_wmma<false, true, true, false><<<gF2, 256, 0, stream>>>(
        hbuf, w216, b2, y1, out, nullptr, M, DM, DF);
}
